// PositionDecoder_7052336300430
// MI455X (gfx1250) — matrix operand order repaired and checked by reading, not compiled, not run on hardware
//
#include <hip/hip_runtime.h>
#include <hip/hip_bf16.h>
#include <math.h>

typedef _Float16 v16h __attribute__((ext_vector_type(16)));
typedef _Float16 v8h  __attribute__((ext_vector_type(8)));
typedef float    v8f  __attribute__((ext_vector_type(8)));

#define N_LAYER   4
#define BSZ       2
#define N_HEADS   16
#define IN_LEN    1024
#define ENC_LEN   588
#define ENC_PAD   608              // 19 * 32
#define HID       256
#define ROWS      (BSZ * IN_LEN)   // 2048

// ---- workspace layout (bytes, all 256B-aligned) ----
#define OFF_XN   ((size_t)0)                               // 2048*608*2  = 2,490,368
#define OFF_WF0  (OFF_XN  + (size_t)ROWS * ENC_PAD * 2)    // 608*256*2   =   311,296
#define OFF_WF1  (OFF_WF0 + (size_t)ENC_PAD * HID * 2)
#define OFF_WF2  (OFF_WF1 + (size_t)ENC_PAD * HID * 2)     // 256*256*2   =   131,072
#define OFF_WF3  (OFF_WF2 + (size_t)HID * HID * 2)
#define OFF_H1   (OFF_WF3 + (size_t)HID * HID * 2)         // 2*2048*256*2 = 2,097,152
#define OFF_H2   (OFF_H1  + (size_t)2 * ROWS * HID * 2)

__device__ __forceinline__ float gelu_exact(float x) {
    return 0.5f * x * (1.0f + erff(x * 0.70710678118654752f));
}

// -------------------------------------------------------------------------
// Pre-swizzle a (Kv x 256) f32 weight matrix into per-lane WMMA B fragments
// (f16, K padded to Kp). Fragment layout for B 32x16 f16 tile:
//   lane<16 : K = kstep*32 + j (j=0..15), lane>=16: K = kstep*32+16+j
//   column  : ntile*16 + (lane&15)
// stored so each lane's 16 halves are contiguous (=> global_load_b128 x2).
// -------------------------------------------------------------------------
__global__ __launch_bounds__(256) void prep_frag_kernel(
    const float* __restrict__ W, _Float16* __restrict__ dst, int Kv, int Kp)
{
    int idx   = blockIdx.x * 256 + threadIdx.x;   // Kp*256 total
    int j     = idx & 15;
    int lane  = (idx >> 4) & 31;
    int ntile = (idx >> 9) & 15;
    int kstep = idx >> 13;
    int k = kstep * 32 + (lane >> 4) * 16 + j;
    int n = ntile * 16 + (lane & 15);
    float v = (k < Kv) ? W[(size_t)k * HID + n] : 0.0f;
    dst[idx] = (_Float16)v;
}

// -------------------------------------------------------------------------
// Stage 1: per (b,t) row: channel mix over 64 (layer,head) weights,
// keep-logit dot, LayerNorm(588) with g0/beta0, store f16 row padded to 608.
// One block (256 threads) per row; coalesced 588-wide streams. Memory-bound:
// 308 MB of heatmap => ~13us floor at 23.3 TB/s.
// -------------------------------------------------------------------------
__global__ __launch_bounds__(256) void head_reduce_kernel(
    const float* __restrict__ hm, const float* __restrict__ W_head,
    const float* __restrict__ b_head, const float* __restrict__ W_keep,
    const float* __restrict__ b_keep, const float* __restrict__ g0,
    const float* __restrict__ beta0, _Float16* __restrict__ xn,
    float* __restrict__ logits)
{
    __shared__ float sW[64];
    __shared__ float red0[256], red1[256], red2[256];

    const int tid = threadIdx.x;
    const int row = blockIdx.x;
    const int b   = row >> 10;
    const int t   = row & 1023;

    if (tid < 64) sW[tid] = W_head[tid];
    __syncthreads();

    const size_t chanStride = (size_t)IN_LEN * ENC_LEN;       // 602112
    const size_t base = (size_t)(b * N_HEADS) * chanStride + (size_t)t * ENC_LEN;

    float xv[3] = {0.f, 0.f, 0.f};
    float s1 = 0.f, s2 = 0.f, sk = 0.f;
    const float bh = b_head[0];

    #pragma unroll
    for (int it = 0; it < 3; ++it) {
        int e = tid + it * 256;
        if (e < ENC_LEN) {
            float acc = 0.f;
            #pragma unroll 4
            for (int c = 0; c < 64; ++c) {
                // channel c = l*16+h maps to physical (l*32 + b*16 + h)
                size_t chan = (size_t)((c >> 4) * 32 + (c & 15)) * chanStride;
                acc = fmaf(sW[c], hm[base + chan + e], acc);
            }
            acc += bh;
            xv[it] = acc;
            s1 += acc;
            s2 = fmaf(acc, acc, s2);
            sk = fmaf(acc, W_keep[e], sk);
        }
    }
    red0[tid] = s1; red1[tid] = s2; red2[tid] = sk;
    __syncthreads();
    for (int off = 128; off > 0; off >>= 1) {
        if (tid < off) {
            red0[tid] += red0[tid + off];
            red1[tid] += red1[tid + off];
            red2[tid] += red2[tid + off];
        }
        __syncthreads();
    }
    const float inv = 1.0f / (float)ENC_LEN;
    float mu   = red0[0] * inv;
    float var  = red1[0] * inv - mu * mu;
    float rstd = rsqrtf(var + 1e-5f);
    if (tid == 0) logits[row] = red2[0] + b_keep[0];

    _Float16* xr = xn + (size_t)row * ENC_PAD;
    #pragma unroll
    for (int it = 0; it < 3; ++it) {
        int e = tid + it * 256;
        if (e < ENC_PAD) {
            float v = (e < ENC_LEN) ? ((xv[it] - mu) * rstd * g0[e] + beta0[e]) : 0.f;
            xr[e] = (_Float16)v;
        }
    }
}

// -------------------------------------------------------------------------
// Fused GEMM + bias + exact gelu + LayerNorm, both branches via blockIdx.y.
// Block: 256 threads = 8 waves; 16-row M block, each wave owns 2 N-tiles
// (8 waves * 2 * 16 = full 256 cols). A staged in LDS; A fragment built per
// contiguous k, lane half h = k 16h..16h+15 (ds_load_b128 x2), B fragments pre-swizzled in ws
// (global_load_b128 x2). K loop of v_wmma_f32_16x16x32_f16.
// -------------------------------------------------------------------------
__global__ __launch_bounds__(256) void gemm_ln_kernel(
    const _Float16* __restrict__ A, size_t aStride,
    const _Float16* __restrict__ Bf0, const _Float16* __restrict__ Bf1,
    const float* __restrict__ bias0, const float* __restrict__ bias1,
    const float* __restrict__ gamma, const float* __restrict__ beta,
    _Float16* __restrict__ out, int nsteps, int Kp)
{
    __shared__ _Float16 sA[16 * ENC_PAD];
    __shared__ float sRes[16 * HID];
    __shared__ float sMu[16], sRstd[16];

    const int tid    = threadIdx.x;
    const int lane   = tid & 31;
    const int wave   = tid >> 5;
    const int br     = blockIdx.y;
    const int rowblk = blockIdx.x * 16;

    const _Float16* Ap  = A + (size_t)br * aStride + (size_t)rowblk * Kp;
    const _Float16* Bf  = br ? Bf1 : Bf0;
    const float*    bia = br ? bias1 : bias0;
    _Float16*       op  = out + (size_t)br * (size_t)ROWS * HID;

    // cooperative load: 16 rows of Kp halves (contiguous block) into LDS
    {
        const uint32_t* src = (const uint32_t*)Ap;
        uint32_t* dst = (uint32_t*)sA;
        const int total = 16 * Kp / 2;
        for (int i = tid; i < total; i += 256) dst[i] = src[i];
    }
    __syncthreads();

    const int half = lane >> 4;
    const int mlo  = lane & 15;
    const int nt0  = wave * 2;
    const int nt1  = nt0 + 1;

    v8f acc0 = {}; v8f acc1 = {};

    for (int ks = 0; ks < nsteps; ++ks) {
        // A fragment: lane<16 -> K {0..15}, lane>=16 -> K {16..31}
        const _Float16* ap = &sA[mlo * Kp + ks * 32 + half * 16];
        v8h alo = *(const v8h*)(ap);
        v8h ahi = *(const v8h*)(ap + 8);
        v16h a = __builtin_shufflevector(alo, ahi,
                 0,1,2,3,4,5,6,7,8,9,10,11,12,13,14,15);

        const _Float16* bp0 = Bf + ((size_t)(ks * 16 + nt0) * 32 + lane) * 16;
        const _Float16* bp1 = Bf + ((size_t)(ks * 16 + nt1) * 32 + lane) * 16;
        v8h b0lo = *(const v8h*)bp0;       v8h b0hi = *(const v8h*)(bp0 + 8);
        v8h b1lo = *(const v8h*)bp1;       v8h b1hi = *(const v8h*)(bp1 + 8);
        v16h bf0 = __builtin_shufflevector(b0lo, b0hi,
                   0,1,2,3,4,5,6,7,8,9,10,11,12,13,14,15);
        v16h bf1 = __builtin_shufflevector(b1lo, b1hi,
                   0,1,2,3,4,5,6,7,8,9,10,11,12,13,14,15);

        acc0 = __builtin_amdgcn_wmma_f32_16x16x32_f16(
                   false, a, false, bf0, (short)0, acc0, false, false);
        acc1 = __builtin_amdgcn_wmma_f32_16x16x32_f16(
                   false, a, false, bf1, (short)0, acc1, false, false);
    }

    // epilogue: bias + exact gelu -> LDS (f32), full 16x256 block resident
    {
        const int col0 = nt0 * 16 + mlo;
        const int col1 = nt1 * 16 + mlo;
        #pragma unroll
        for (int r = 0; r < 8; ++r) {
            int m = half * 8 + r;        // C/D layout: VGPR r -> M = half*8 + r
            sRes[m * HID + col0] = gelu_exact(acc0[r] + bia[col0]);
            sRes[m * HID + col1] = gelu_exact(acc1[r] + bia[col1]);
        }
    }
    __syncthreads();
    if (tid < 16) {
        float s = 0.f, s2 = 0.f;
        for (int c = 0; c < HID; ++c) {
            float v = sRes[tid * HID + c];
            s += v; s2 = fmaf(v, v, s2);
        }
        float mu  = s * (1.0f / HID);
        float var = s2 * (1.0f / HID) - mu * mu;
        sMu[tid]   = mu;
        sRstd[tid] = rsqrtf(var + 1e-5f);
    }
    __syncthreads();
    for (int i = tid; i < 16 * HID; i += 256) {
        int m = i >> 8, c = i & 255;
        float v = (sRes[i] - sMu[m]) * sRstd[m] * gamma[c] + beta[c];
        op[(size_t)(rowblk + m) * HID + c] = (_Float16)v;
    }
}

// -------------------------------------------------------------------------
// Final: pick branch by keep logit, 256->4 matvec, sigmoid, mask, boxes.
// -------------------------------------------------------------------------
__global__ __launch_bounds__(256) void final_kernel(
    const _Float16* __restrict__ h2,
    const float* __restrict__ W1_2, const float* __restrict__ b1_2,
    const float* __restrict__ W2_2, const float* __restrict__ b2_2,
    const float* __restrict__ avm, float* __restrict__ out)
{
    int r = blockIdx.x * 256 + threadIdx.x;
    if (r >= ROWS) return;
    float logit = out[4 * ROWS + r];
    int   br    = (logit > 0.f) ? 0 : 1;         // sigmoid(x)>0.5 <=> x>0
    const _Float16* hrow = h2 + ((size_t)br * ROWS + r) * HID;
    const float* W  = br ? W2_2 : W1_2;
    const float* bb = br ? b2_2 : b1_2;

    float a0 = bb[0], a1 = bb[1], a2 = bb[2], a3 = bb[3];
    for (int k = 0; k < HID; ++k) {
        float v = (float)hrow[k];
        const float* w = W + k * 4;
        a0 = fmaf(v, w[0], a0);
        a1 = fmaf(v, w[1], a1);
        a2 = fmaf(v, w[2], a2);
        a3 = fmaf(v, w[3], a3);
    }
    float m  = avm[r];
    float o0 = m / (1.f + expf(-a0));
    float o1 = m / (1.f + expf(-a1));
    float o2 = m / (1.f + expf(-a2));
    float o3 = m / (1.f + expf(-a3));
    out[r]            = o0;
    out[ROWS + r]     = o1;
    out[2 * ROWS + r] = o0 + o2;
    out[3 * ROWS + r] = o1 + o3;
}

extern "C" void kernel_launch(void* const* d_in, const int* in_sizes, int n_in,
                              void* d_out, int out_size, void* d_ws, size_t ws_size,
                              hipStream_t stream) {
    (void)in_sizes; (void)n_in; (void)out_size; (void)ws_size;

    const float* hm    = (const float*)d_in[0];
    const float* avm   = (const float*)d_in[1];
    const float* Whead = (const float*)d_in[2];
    const float* bhead = (const float*)d_in[3];
    const float* Wkeep = (const float*)d_in[4];
    const float* bkeep = (const float*)d_in[5];
    const float* W1_0  = (const float*)d_in[6];
    const float* b1_0  = (const float*)d_in[7];
    const float* W1_1  = (const float*)d_in[8];
    const float* b1_1  = (const float*)d_in[9];
    const float* W1_2  = (const float*)d_in[10];
    const float* b1_2  = (const float*)d_in[11];
    const float* W2_0  = (const float*)d_in[12];
    const float* b2_0  = (const float*)d_in[13];
    const float* W2_1  = (const float*)d_in[14];
    const float* b2_1  = (const float*)d_in[15];
    const float* W2_2  = (const float*)d_in[16];
    const float* b2_2  = (const float*)d_in[17];
    const float* g0    = (const float*)d_in[18];
    const float* be0   = (const float*)d_in[19];
    const float* g1    = (const float*)d_in[20];
    const float* be1   = (const float*)d_in[21];
    const float* g2    = (const float*)d_in[22];
    const float* be2   = (const float*)d_in[23];

    char* ws = (char*)d_ws;
    _Float16* xn  = (_Float16*)(ws + OFF_XN);
    _Float16* wf0 = (_Float16*)(ws + OFF_WF0);
    _Float16* wf1 = (_Float16*)(ws + OFF_WF1);
    _Float16* wf2 = (_Float16*)(ws + OFF_WF2);
    _Float16* wf3 = (_Float16*)(ws + OFF_WF3);
    _Float16* h1  = (_Float16*)(ws + OFF_H1);
    _Float16* h2  = (_Float16*)(ws + OFF_H2);
    float*    out = (float*)d_out;

    // weight pre-swizzle into WMMA B-fragment layout (f16, K zero-padded)
    prep_frag_kernel<<<ENC_PAD, 256, 0, stream>>>(W1_0, wf0, ENC_LEN, ENC_PAD);
    prep_frag_kernel<<<ENC_PAD, 256, 0, stream>>>(W2_0, wf1, ENC_LEN, ENC_PAD);
    prep_frag_kernel<<<HID,     256, 0, stream>>>(W1_1, wf2, HID, HID);
    prep_frag_kernel<<<HID,     256, 0, stream>>>(W2_1, wf3, HID, HID);

    // stage 1: channel mix + keep logit + LN0 (memory-bound, ~13us floor)
    head_reduce_kernel<<<ROWS, 256, 0, stream>>>(
        hm, Whead, bhead, Wkeep, bkeep, g0, be0, xn, out);

    // layer 0: xn(2048x608 f16) @ W*_0 -> gelu -> LN1 -> h1 (both branches)
    gemm_ln_kernel<<<dim3(ROWS / 16, 2), 256, 0, stream>>>(
        xn, (size_t)0, wf0, wf1, b1_0, b2_0, g1, be1, h1, ENC_PAD / 32, ENC_PAD);

    // layer 1: h1 @ W*_1 -> gelu -> LN2 -> h2 (both branches)
    gemm_ln_kernel<<<dim3(ROWS / 16, 2), 256, 0, stream>>>(
        h1, (size_t)ROWS * HID, wf2, wf3, b1_1, b2_1, g2, be2, h2, HID / 32, HID);

    // final: branch select + 256->4 + sigmoid + mask + box assembly
    final_kernel<<<ROWS / 256, 256, 0, stream>>>(
        h2, W1_2, b1_2, W2_2, b2_2, avm, out);
}
